// EGNNLayer_53979148976479
// MI455X (gfx1250) — matrix operand order repaired and checked by reading, compiled, not run on hardware
//
#include <hip/hip_runtime.h>

// ---------------------------------------------------------------------------
// EGNN layer for MI455X (gfx1250, wave32, WMMA bf16 16x16x32 f32-accum).
// GEMMs run on v_wmma_f32_16x16x32_bf16. Pre-transposed bf16 weights are
// staged into LDS once per workgroup (8 waves share them), so B-fragments
// come from ds_load_b128 instead of per-WMMA L2 round trips. Layer-1
// activations are staged through LDS (bf16) to rebuild the A-fragment
// layout for layer 2.
// ---------------------------------------------------------------------------

typedef __attribute__((ext_vector_type(16))) __bf16 v16bf;
typedef __attribute__((ext_vector_type(8)))  float  v8f;

#define KS1  260   // k-stride for 257-row (W_v1/W_a1) and 256-row (W_h1) transposed weights
#define KS2  132   // k-stride for 128-row transposed weights (W_v2/W_h2)
#define ACTS 136   // LDS activation staging stride (bf16 elements)

__device__ __forceinline__ float siluf(float x) {
    return x / (1.0f + __expf(-x));
}
// monotonic order-preserving float->uint encoding for atomicMax
__device__ __forceinline__ unsigned fenc(float f) {
    int i = __float_as_int(f);
    return (i >= 0) ? (unsigned(i) ^ 0x80000000u) : ~unsigned(i);
}
__device__ __forceinline__ float fdec(unsigned u) {
    int i = (u & 0x80000000u) ? int(u ^ 0x80000000u) : ~int(u);
    return __int_as_float(i);
}
__device__ __forceinline__ v8f vzero8() {
    v8f z;
#pragma unroll
    for (int i = 0; i < 8; ++i) z[i] = 0.0f;
    return z;
}
// A-fragment (16x32 bf16, M = lane%16): p points at row base + kt*32 + (hi?8:0)
__device__ __forceinline__ v16bf load_a_f32(const float* p, float scale) {
    v16bf a;
#pragma unroll
    for (int j = 0; j < 8; ++j)  a[j] = (__bf16)(p[j] * scale);
#pragma unroll
    for (int j = 8; j < 16; ++j) a[j] = (__bf16)(p[j + 8] * scale);
    return a;
}
__device__ __forceinline__ v16bf load_a_bf16(const __bf16* p) {
    v16bf a;
#pragma unroll
    for (int j = 0; j < 8; ++j)  a[j] = p[j];
#pragma unroll
    for (int j = 8; j < 16; ++j) a[j] = p[j + 8];
    return a;
}
// B-fragment (32x16 bf16, N = lane%16): p = Wt + n*KS + kbase + (hi?8:0)
__device__ __forceinline__ v16bf load_b(const __bf16* p) {
    v16bf b;
#pragma unroll
    for (int j = 0; j < 8; ++j)  b[j] = p[j];
#pragma unroll
    for (int j = 8; j < 16; ++j) b[j] = p[j + 8];
    return b;
}
__device__ __forceinline__ v8f wmma_bf16(v16bf a, v16bf b, v8f c) {
    return __builtin_amdgcn_wmma_f32_16x16x32_bf16(false, a, false, b,
                                                   (short)0, c, false, false);
}
// cooperative global(bf16)->LDS copy, 256 threads, 32-bit words
__device__ __forceinline__ void stage_weights(__bf16* dst, const __bf16* src,
                                              int n_bf16) {
    const unsigned* g = (const unsigned*)src;
    unsigned* l = (unsigned*)dst;
    const int words = n_bf16 >> 1;
    for (int i = threadIdx.x; i < words; i += 256) l[i] = g[i];
}

// ---------------------------------------------------------------------------
// Weight pre-transpose: Wt[n][k] = (bf16) W[k][n], n in [0,128)
// ---------------------------------------------------------------------------
__global__ void k_transpose(const float* __restrict__ W, __bf16* __restrict__ Wt,
                            int KS, int total) {
    int idx = blockIdx.x * 256 + threadIdx.x;
    if (idx >= total) return;
    int n = idx & 127;
    int k = idx >> 7;
    Wt[(size_t)n * KS + k] = (__bf16)W[(size_t)k * 128 + n];
}

// ---------------------------------------------------------------------------
// Edge attention: dij/rij, logits = silu(e_in@W_a1+b)@W_a2+b, atomicMax per dst
// One wave = one 16-edge tile. 64 WMMA per wave. W_a1^T staged in LDS.
// ---------------------------------------------------------------------------
__global__ __launch_bounds__(256) void k_edge_attn(
    const float* __restrict__ x_h, const float* __restrict__ x_pos,
    const int* __restrict__ src, const int* __restrict__ dst,
    const __bf16* __restrict__ Wt_a1, const float* __restrict__ b_a1,
    const float* __restrict__ W_a2, const float* __restrict__ b_a2,
    float* __restrict__ logits, float* __restrict__ dij_g,
    float* __restrict__ rij_g, unsigned* __restrict__ maxenc, int E, int NT)
{
    __shared__ __bf16 sW1[128 * KS1];     // 66,560 B
    __shared__ float  s_dij[8][16];
    const int lane = threadIdx.x & 31;
    const int wave = threadIdx.x >> 5;
    const int l16  = lane & 15;
    const int hi   = lane >> 4;
    int tile = blockIdx.x * 8 + wave;
    if (tile >= NT) tile = NT - 1;            // duplicate work is idempotent
    const int ebase = tile * 16;

    stage_weights(sW1, Wt_a1, 128 * KS1);

    if (lane < 16) {                          // geometry, one edge per lane
        int e = min(ebase + lane, E - 1);
        int s = src[e], d = dst[e];
        float rx = x_pos[d * 3 + 0] - x_pos[s * 3 + 0];
        float ry = x_pos[d * 3 + 1] - x_pos[s * 3 + 1];
        float rz = x_pos[d * 3 + 2] - x_pos[s * 3 + 2];
        float dd = sqrtf(rx * rx + ry * ry + rz * rz);
        rij_g[e * 3 + 0] = rx; rij_g[e * 3 + 1] = ry; rij_g[e * 3 + 2] = rz;
        dij_g[e] = dd;
        s_dij[wave][lane] = dd;
    }

    const int e_m = min(ebase + l16, E - 1);
    const int sm = src[e_m], dm = dst[e_m];
    v16bf a_src[4], a_dst[4];
#pragma unroll
    for (int kt = 0; kt < 4; ++kt) {
        a_src[kt] = load_a_f32(x_h + (size_t)sm * 128 + kt * 32 + hi * 8, 1.0f);
        a_dst[kt] = load_a_f32(x_h + (size_t)dm * 128 + kt * 32 + hi * 8, 1.0f);
    }
    __syncthreads();

    float logacc[8];
#pragma unroll
    for (int v = 0; v < 8; ++v) logacc[v] = 0.0f;

#pragma unroll
    for (int j = 0; j < 8; ++j) {
        int n = j * 16 + l16;
        v8f acc = vzero8();
#pragma unroll
        for (int kt = 0; kt < 4; ++kt)
            acc = wmma_bf16(a_src[kt], load_b(sW1 + n * KS1 + kt * 32 + hi * 8), acc);
#pragma unroll
        for (int kt = 0; kt < 4; ++kt)
            acc = wmma_bf16(a_dst[kt], load_b(sW1 + n * KS1 + 128 + kt * 32 + hi * 8), acc);
        float w256 = (float)sW1[n * KS1 + 256];  // dist column
        float bn   = b_a1[n];
        float wa2n = W_a2[n];
#pragma unroll
        for (int v = 0; v < 8; ++v) {
            float pre = acc[v] + bn + s_dij[wave][v + hi * 8] * w256;
            logacc[v] += siluf(pre) * wa2n;               // fold layer-2 dot
        }
    }
    // reduce partial dot across the 16 lanes of each half-wave
#pragma unroll
    for (int v = 0; v < 8; ++v) {
#pragma unroll
        for (int msk = 1; msk < 16; msk <<= 1)
            logacc[v] += __shfl_xor(logacc[v], msk, 16);
    }
    if (l16 == 0) {
        float ba2 = b_a2[0];
#pragma unroll
        for (int v = 0; v < 8; ++v) {
            int e = ebase + v + hi * 8;
            if (e < E) {
                float lg = logacc[v] + ba2;
                logits[e] = lg;
                atomicMax(&maxenc[dst[e]], fenc(lg));
            }
        }
    }
}

// ---------------------------------------------------------------------------
// Edge value MLP: v_ij = silu(silu(e_in@W_v1+b)@W_v2+b) -> bf16. 96 WMMA/wave.
// W_v1^T and W_v2^T staged in LDS.
// ---------------------------------------------------------------------------
__global__ __launch_bounds__(256) void k_edge_value(
    const float* __restrict__ x_h,
    const int* __restrict__ src, const int* __restrict__ dst,
    const __bf16* __restrict__ Wt_v1, const float* __restrict__ b_v1,
    const __bf16* __restrict__ Wt_v2, const float* __restrict__ b_v2,
    const float* __restrict__ dij_g, __bf16* __restrict__ vij, int E, int NT)
{
    __shared__ __bf16 sW1[128 * KS1];        // 66,560 B
    __shared__ __bf16 sW2[128 * KS2];        // 33,792 B
    __shared__ __bf16 s_act[8][16][ACTS];    // 34,816 B
    __shared__ float  s_dij[8][16];
    const int lane = threadIdx.x & 31;
    const int wave = threadIdx.x >> 5;
    const int l16  = lane & 15;
    const int hi   = lane >> 4;
    int tile = blockIdx.x * 8 + wave;
    if (tile >= NT) tile = NT - 1;
    const int ebase = tile * 16;

    stage_weights(sW1, Wt_v1, 128 * KS1);
    stage_weights(sW2, Wt_v2, 128 * KS2);

    if (lane < 16) s_dij[wave][lane] = dij_g[min(ebase + lane, E - 1)];

    const int e_m = min(ebase + l16, E - 1);
    const int sm = src[e_m], dm = dst[e_m];
    v16bf a_src[4], a_dst[4];
#pragma unroll
    for (int kt = 0; kt < 4; ++kt) {
        a_src[kt] = load_a_f32(x_h + (size_t)sm * 128 + kt * 32 + hi * 8, 1.0f);
        a_dst[kt] = load_a_f32(x_h + (size_t)dm * 128 + kt * 32 + hi * 8, 1.0f);
    }
    __syncthreads();

#pragma unroll
    for (int j = 0; j < 8; ++j) {
        int n = j * 16 + l16;
        v8f acc = vzero8();
#pragma unroll
        for (int kt = 0; kt < 4; ++kt)
            acc = wmma_bf16(a_src[kt], load_b(sW1 + n * KS1 + kt * 32 + hi * 8), acc);
#pragma unroll
        for (int kt = 0; kt < 4; ++kt)
            acc = wmma_bf16(a_dst[kt], load_b(sW1 + n * KS1 + 128 + kt * 32 + hi * 8), acc);
        float w256 = (float)sW1[n * KS1 + 256];
        float bn   = b_v1[n];
#pragma unroll
        for (int v = 0; v < 8; ++v) {   // C layout -> LDS [m][n]
            float pre = acc[v] + bn + s_dij[wave][v + hi * 8] * w256;
            s_act[wave][v + hi * 8][n] = (__bf16)siluf(pre);
        }
    }
    __syncthreads();

    v16bf a2[4];
#pragma unroll
    for (int kt = 0; kt < 4; ++kt)
        a2[kt] = load_a_bf16(&s_act[wave][l16][kt * 32 + hi * 8]);

#pragma unroll
    for (int j2 = 0; j2 < 8; ++j2) {
        int n2 = j2 * 16 + l16;
        v8f acc = vzero8();
#pragma unroll
        for (int kt = 0; kt < 4; ++kt)
            acc = wmma_bf16(a2[kt], load_b(sW2 + n2 * KS2 + kt * 32 + hi * 8), acc);
        float bn = b_v2[n2];
#pragma unroll
        for (int v = 0; v < 8; ++v) {
            int e = ebase + v + hi * 8;
            if (e < E) vij[(size_t)e * 128 + n2] = (__bf16)siluf(acc[v] + bn);
        }
    }
}

// ---------------------------------------------------------------------------
// exp(logit - max[dst]) + denom + degree
// ---------------------------------------------------------------------------
__global__ void k_expw(const int* __restrict__ dst, const float* __restrict__ logits,
                       const unsigned* __restrict__ maxenc, float* __restrict__ wexp,
                       float* __restrict__ denom, float* __restrict__ deg, int E)
{
    int e = blockIdx.x * 256 + threadIdx.x;
    if (e >= E) return;
    int d = dst[e];
    float w = __expf(logits[e] - fdec(maxenc[d]));
    wexp[e] = w;
    atomicAdd(&denom[d], w);
    atomicAdd(&deg[d], 1.0f);
}

// ---------------------------------------------------------------------------
// Per-edge scatter: m_ij = v_ij*alpha; h_agg += m_ij; x_agg += rij*(m_ij@W_x)
// One wave per edge (4 dims per lane).
// ---------------------------------------------------------------------------
__global__ __launch_bounds__(256) void k_scatter(
    const int* __restrict__ dst, const float* __restrict__ wexp,
    const float* __restrict__ denom, const __bf16* __restrict__ vij,
    const float* __restrict__ W_x, const float* __restrict__ rij_g,
    float* __restrict__ hagg, float* __restrict__ xagg, int E)
{
    const int wave = threadIdx.x >> 5;
    const int lane = threadIdx.x & 31;
    int e = blockIdx.x * 8 + wave;
    if (e >= E) return;
    int d = dst[e];
    float alpha = wexp[e] / (denom[d] + 1e-9f);
    float coord = 0.0f;
#pragma unroll
    for (int u = 0; u < 4; ++u) {
        int idx = lane * 4 + u;
        float mv = (float)vij[(size_t)e * 128 + idx] * alpha;
        atomicAdd(&hagg[(size_t)d * 128 + idx], mv);
        coord += mv * W_x[idx];
    }
#pragma unroll
    for (int msk = 1; msk < 32; msk <<= 1)
        coord += __shfl_xor(coord, msk, 32);
    if (lane < 3)
        atomicAdd(&xagg[d * 3 + lane], rij_g[e * 3 + lane] * coord);
}

// ---------------------------------------------------------------------------
// Node MLP: h_out = silu([x_h, h_agg/deg]@W_h1+b)@W_h2+b + x_h. 96 WMMA/wave.
// W_h1^T and W_h2^T staged in LDS.
// ---------------------------------------------------------------------------
__global__ __launch_bounds__(256) void k_node(
    const float* __restrict__ x_h, const float* __restrict__ hagg,
    const float* __restrict__ deg,
    const __bf16* __restrict__ Wt_h1, const float* __restrict__ b_h1,
    const __bf16* __restrict__ Wt_h2, const float* __restrict__ b_h2,
    float* __restrict__ h_out, int N, int NT)
{
    __shared__ __bf16 sW1[128 * KS1];        // 66,560 B
    __shared__ __bf16 sW2[128 * KS2];        // 33,792 B
    __shared__ __bf16 s_act[8][16][ACTS];    // 34,816 B
    const int lane = threadIdx.x & 31;
    const int wave = threadIdx.x >> 5;
    const int l16  = lane & 15;
    const int hi   = lane >> 4;
    int tile = blockIdx.x * 8 + wave;
    if (tile >= NT) tile = NT - 1;
    const int nbase = tile * 16;

    stage_weights(sW1, Wt_h1, 128 * KS1);
    stage_weights(sW2, Wt_h2, 128 * KS2);

    const int row = min(nbase + l16, N - 1);
    float invd = 1.0f / fmaxf(deg[row], 1.0f);
    v16bf a_x[4], a_g[4];
#pragma unroll
    for (int kt = 0; kt < 4; ++kt) {
        a_x[kt] = load_a_f32(x_h  + (size_t)row * 128 + kt * 32 + hi * 8, 1.0f);
        a_g[kt] = load_a_f32(hagg + (size_t)row * 128 + kt * 32 + hi * 8, invd);
    }
    __syncthreads();

#pragma unroll
    for (int j = 0; j < 8; ++j) {
        int n = j * 16 + l16;
        v8f acc = vzero8();
#pragma unroll
        for (int kt = 0; kt < 4; ++kt)
            acc = wmma_bf16(a_x[kt], load_b(sW1 + n * KS1 + kt * 32 + hi * 8), acc);
#pragma unroll
        for (int kt = 0; kt < 4; ++kt)
            acc = wmma_bf16(a_g[kt], load_b(sW1 + n * KS1 + 128 + kt * 32 + hi * 8), acc);
        float bn = b_h1[n];
#pragma unroll
        for (int v = 0; v < 8; ++v)
            s_act[wave][v + hi * 8][n] = (__bf16)siluf(acc[v] + bn);
    }
    __syncthreads();

    v16bf a2[4];
#pragma unroll
    for (int kt = 0; kt < 4; ++kt)
        a2[kt] = load_a_bf16(&s_act[wave][l16][kt * 32 + hi * 8]);

#pragma unroll
    for (int j2 = 0; j2 < 8; ++j2) {
        int n2 = j2 * 16 + l16;
        v8f acc = vzero8();
#pragma unroll
        for (int kt = 0; kt < 4; ++kt)
            acc = wmma_bf16(a2[kt], load_b(sW2 + n2 * KS2 + kt * 32 + hi * 8), acc);
        float bn = b_h2[n2];
#pragma unroll
        for (int v = 0; v < 8; ++v) {
            int r = nbase + v + hi * 8;
            if (r < N)
                h_out[(size_t)r * 128 + n2] = acc[v] + bn + x_h[(size_t)r * 128 + n2];
        }
    }
}

// ---------------------------------------------------------------------------
// x_out = x_pos + x_agg/deg
// ---------------------------------------------------------------------------
__global__ void k_xout(const float* __restrict__ x_pos, const float* __restrict__ xagg,
                       const float* __restrict__ deg, float* __restrict__ x_out, int N)
{
    int i = blockIdx.x * 256 + threadIdx.x;
    if (i >= N * 3) return;
    int node = i / 3;
    x_out[i] = x_pos[i] + xagg[i] / fmaxf(deg[node], 1.0f);
}

// ---------------------------------------------------------------------------
extern "C" void kernel_launch(void* const* d_in, const int* in_sizes, int n_in,
                              void* d_out, int out_size, void* d_ws, size_t ws_size,
                              hipStream_t stream)
{
    const float* x_h   = (const float*)d_in[0];
    const float* x_pos = (const float*)d_in[1];
    const int*   ei    = (const int*)d_in[2];
    const float* W_v1  = (const float*)d_in[3];
    const float* b_v1  = (const float*)d_in[4];
    const float* W_v2  = (const float*)d_in[5];
    const float* b_v2  = (const float*)d_in[6];
    const float* W_a1  = (const float*)d_in[7];
    const float* b_a1  = (const float*)d_in[8];
    const float* W_a2  = (const float*)d_in[9];
    const float* b_a2  = (const float*)d_in[10];
    const float* W_x   = (const float*)d_in[11];
    const float* W_h1  = (const float*)d_in[12];
    const float* b_h1  = (const float*)d_in[13];
    const float* W_h2  = (const float*)d_in[14];
    const float* b_h2  = (const float*)d_in[15];

    const int N = in_sizes[0] / 128;
    const int E = in_sizes[2] / 2;
    const int* src = ei;
    const int* dst = ei + E;

    // ---- workspace carve-up (256B aligned regions) ----
    char* w = (char*)d_ws;
    size_t off = 0;
    auto carve = [&](size_t bytes) {
        size_t r = off;
        off = (off + bytes + 255) & ~(size_t)255;
        return r;
    };
    unsigned* maxenc = (unsigned*)(w + carve((size_t)N * 4));
    float*    denom  = (float*)(w + carve((size_t)N * 4));
    float*    deg    = (float*)(w + carve((size_t)N * 4));
    float*    hagg   = (float*)(w + carve((size_t)N * 128 * 4));
    float*    xagg   = (float*)(w + carve((size_t)N * 3 * 4));
    float*    logits = (float*)(w + carve((size_t)E * 4));
    float*    wexp   = (float*)(w + carve((size_t)E * 4));
    float*    dij    = (float*)(w + carve((size_t)E * 4));
    float*    rij    = (float*)(w + carve((size_t)E * 3 * 4));
    __bf16*   vij    = (__bf16*)(w + carve((size_t)E * 128 * 2));
    __bf16*   Wt_v1  = (__bf16*)(w + carve((size_t)128 * KS1 * 2));
    __bf16*   Wt_a1  = (__bf16*)(w + carve((size_t)128 * KS1 * 2));
    __bf16*   Wt_h1  = (__bf16*)(w + carve((size_t)128 * KS1 * 2));
    __bf16*   Wt_v2  = (__bf16*)(w + carve((size_t)128 * KS2 * 2));
    __bf16*   Wt_h2  = (__bf16*)(w + carve((size_t)128 * KS2 * 2));

    hipMemsetAsync(maxenc, 0, (size_t)N * 4, stream);         // 0 == -inf encoding floor
    hipMemsetAsync(denom,  0, (size_t)N * 4, stream);
    hipMemsetAsync(deg,    0, (size_t)N * 4, stream);
    hipMemsetAsync(hagg,   0, (size_t)N * 128 * 4, stream);
    hipMemsetAsync(xagg,   0, (size_t)N * 3 * 4, stream);

    auto launchT = [&](const float* W, __bf16* Wt, int K, int KS) {
        int total = 128 * K;
        k_transpose<<<(total + 255) / 256, 256, 0, stream>>>(W, Wt, KS, total);
    };
    launchT(W_v1, Wt_v1, 257, KS1);
    launchT(W_a1, Wt_a1, 257, KS1);
    launchT(W_h1, Wt_h1, 256, KS1);
    launchT(W_v2, Wt_v2, 128, KS2);
    launchT(W_h2, Wt_h2, 128, KS2);

    const int NT_E = (E + 15) / 16;
    const int blkE = (NT_E + 7) / 8;
    k_edge_attn<<<blkE, 256, 0, stream>>>(x_h, x_pos, src, dst, Wt_a1, b_a1,
                                          W_a2, b_a2, logits, dij, rij, maxenc, E, NT_E);
    k_edge_value<<<blkE, 256, 0, stream>>>(x_h, src, dst, Wt_v1, b_v1,
                                           Wt_v2, b_v2, dij, vij, E, NT_E);
    k_expw<<<(E + 255) / 256, 256, 0, stream>>>(dst, logits, maxenc, wexp, denom, deg, E);
    k_scatter<<<(E + 7) / 8, 256, 0, stream>>>(dst, wexp, denom, vij, W_x, rij, hagg, xagg, E);

    const int NT_N = (N + 15) / 16;
    k_node<<<(NT_N + 7) / 8, 256, 0, stream>>>(x_h, hagg, deg, Wt_h1, b_h1,
                                               Wt_h2, b_h2, (float*)d_out, N, NT_N);
    k_xout<<<(N * 3 + 255) / 256, 256, 0, stream>>>(x_pos, xagg, deg,
                                                    (float*)d_out + (size_t)N * 128, N);
}
